// Model_30872224923863
// MI455X (gfx1250) — matrix operand order repaired and checked by reading, not compiled, not run on hardware
//
#include <hip/hip_runtime.h>
#include <hip/hip_bf16.h>

// ---------------------------------------------------------------------------
// Seq2seq GRU + attention for MI455X (gfx1250, wave32, WMMA).
// GEMMs: v_wmma_f32_16x16x32_f16, 8-wave blocks, 128x64 tiles.
// B operand staged through double-buffered LDS (async-to-LDS via
// global_load_async_to_lds_b128, ASYNCcnt-tracked); A operand software-
// pipelined in registers so loadcnt waits overlap WMMA issue.
// Weights packed once per launch to N-major f16 (~9.4MB, L2-resident).
// ---------------------------------------------------------------------------

typedef _Float16 h4   __attribute__((ext_vector_type(4)));
typedef _Float16 h8   __attribute__((ext_vector_type(8)));
typedef _Float16 v16h __attribute__((ext_vector_type(16)));
typedef float    f4   __attribute__((ext_vector_type(4)));
typedef float    v8f  __attribute__((ext_vector_type(8)));

// generic-vector int4 exactly as the async-LDS builtin's prototype spells it
typedef int i4v __attribute__((vector_size(16)));
typedef __attribute__((address_space(1))) i4v* gv4p;
typedef __attribute__((address_space(3))) i4v* lv4p;

#define B_   2048
#define S_   32
#define T_   32
#define D_   256
#define U_   512
#define U3_  1536
#define VT_  256
#define BOW_ 1

union V16U { v16h v; h8 o[2]; h4 q[4]; };

#if __has_builtin(__builtin_amdgcn_global_load_async_to_lds_b128) && \
    __has_builtin(__builtin_amdgcn_s_wait_asynccnt)
#define ASYNC_LDS 1
#else
#define ASYNC_LDS 0
#endif

// ---------------- weight pack: f32 [K,N] -> f16 [N,K] ----------------------
__global__ void k_pack_weight(const float* __restrict__ W, _Float16* __restrict__ out,
                              int K, int N) {
  int idx = blockIdx.x * blockDim.x + threadIdx.x;
  if (idx >= K * N) return;
  int n = idx / K, k = idx - n * K;
  out[idx] = (_Float16)W[(long)k * N + n];
}

// ---------------- gather source embeddings into [S,B,D] --------------------
__global__ void k_gather_src(const int* __restrict__ ids, const float* __restrict__ Es,
                             float* __restrict__ xs) {
  int idx = blockIdx.x * blockDim.x + threadIdx.x;   // B*S*(D/4)
  int d4 = idx & (D_ / 4 - 1);
  int bs = idx / (D_ / 4);
  int s = bs & (S_ - 1);
  int b = bs / S_;
  int tok = ids[b * S_ + s];
  ((f4*)xs)[(long)(s * B_ + b) * (D_ / 4) + d4] =
      ((const f4*)Es)[(long)tok * (D_ / 4) + d4];
}

// ---------------- gather target embedding into inp[:, 0:D] -----------------
__global__ void k_gather_tgt(const int* __restrict__ ids, int col, int fixed_id,
                             const float* __restrict__ Et, float* __restrict__ inp) {
  int idx = blockIdx.x * blockDim.x + threadIdx.x;   // B*(D/4)
  int d4 = idx & (D_ / 4 - 1);
  int b = idx / (D_ / 4);
  int tok = (fixed_id >= 0) ? fixed_id : ids[b * T_ + col];
  ((f4*)inp)[(long)b * ((D_ + U_) / 4) + d4] =
      ((const f4*)Et)[(long)tok * (D_ / 4) + d4];
}

__global__ void k_zero(float* __restrict__ p, int n) {
  int i = blockIdx.x * blockDim.x + threadIdx.x;
  if (i < n) p[i] = 0.0f;
}

// ---------------- 16-byte global -> LDS stage ------------------------------
__device__ __forceinline__ void stage16(const _Float16* __restrict__ g,
                                        _Float16* l) {
#if ASYNC_LDS
  _Float16* gnc = const_cast<_Float16*>(g);
  __builtin_amdgcn_global_load_async_to_lds_b128(
      (gv4p)(__attribute__((address_space(1))) void*)gnc,
      (lv4p)(__attribute__((address_space(3))) void*)l, 0, 0);
#else
  *(f4*)l = *(const f4*)g;
#endif
}

// ---------------- WMMA GEMM: C[M,N] = A[M,K](f32) * Bp[N,K](f16) + bias ----
// grid = (N/64, M/128), block = 256 threads = 8 waves; wave -> 16 rows x 64 cols.
// B slice (64 n x 32 k) staged in double-buffered LDS shared by all 8 waves.
__global__ __launch_bounds__(256) void k_wmma_gemm(
    const float* __restrict__ A, int lda,
    const _Float16* __restrict__ Bp, int K,
    const float* __restrict__ bias,
    float* __restrict__ C, long ldc) {
  __shared__ __attribute__((aligned(16))) _Float16 sB[2][64 * 32];

  const int tid  = threadIdx.x;
  const int lane = tid & 31;
  const int wave = tid >> 5;
  const int l15  = lane & 15;
  const int kg   = lane >> 4;                   // 0 or 1
  const long mRow  = (long)blockIdx.y * 128 + wave * 16 + l15;
  const int  nBase = blockIdx.x * 64;

  // cooperative B staging: thread -> (row = tid>>2, 16B segment = tid&3)
  const int sRow = tid >> 2;
  const int sSeg = (tid & 3) * 8;
  const _Float16* bStageG = Bp + (long)(nBase + sRow) * K + sSeg;
  _Float16* bStageL0 = &sB[0][sRow * 32 + sSeg];
  _Float16* bStageL1 = &sB[1][sRow * 32 + sSeg];

  // A fragment source: lane holds A[m][kg*8 .. +7] and A[m][16+kg*8 .. +7]
  const float* aPtr = A + mRow * lda + kg * 8;

  v8f acc0 = {}, acc1 = {}, acc2 = {}, acc3 = {};

  // ---- prologue: stage B slice 0, load A slice 0 --------------------------
  stage16(bStageG, bStageL0);
  f4 ra0 = *(const f4*)(aPtr + 0);
  f4 ra1 = *(const f4*)(aPtr + 4);
  f4 ra2 = *(const f4*)(aPtr + 16);
  f4 ra3 = *(const f4*)(aPtr + 20);
#if ASYNC_LDS
  __builtin_amdgcn_s_wait_asynccnt(0);
#endif
  __syncthreads();

  for (int k0 = 0; k0 < K; k0 += 32) {
    const int cur = (k0 >> 5) & 1;
    const bool more = (k0 + 32) < K;

    // issue next B stage (async; overlaps with this iteration's compute)
    if (more) stage16(bStageG + k0 + 32, cur ? bStageL0 : bStageL1);

    // issue next A loads (raw f32; converted only next iteration)
    f4 rb0, rb1, rb2, rb3;
    if (more) {
      rb0 = *(const f4*)(aPtr + k0 + 32);
      rb1 = *(const f4*)(aPtr + k0 + 36);
      rb2 = *(const f4*)(aPtr + k0 + 48);
      rb3 = *(const f4*)(aPtr + k0 + 52);
    }

    // convert current A fragment (loaded last iteration -> no wait here)
    V16U af;
    af.q[0] = __builtin_convertvector(ra0, h4);
    af.q[1] = __builtin_convertvector(ra1, h4);
    af.q[2] = __builtin_convertvector(ra2, h4);
    af.q[3] = __builtin_convertvector(ra3, h4);

    // B fragments from LDS: lane holds Bp[n][kg*8 .. +7] and Bp[n][16+kg*8 .. +7]
    const _Float16* sb = &sB[cur][l15 * 32 + kg * 8];
    {
      V16U bf;
      bf.o[0] = *(const h8*)(sb + 0 * 16 * 32);
      bf.o[1] = *(const h8*)(sb + 0 * 16 * 32 + 16);
      acc0 = __builtin_amdgcn_wmma_f32_16x16x32_f16(false, af.v, false, bf.v, (short)0, acc0, false, false);
    }
    {
      V16U bf;
      bf.o[0] = *(const h8*)(sb + 1 * 16 * 32);
      bf.o[1] = *(const h8*)(sb + 1 * 16 * 32 + 16);
      acc1 = __builtin_amdgcn_wmma_f32_16x16x32_f16(false, af.v, false, bf.v, (short)0, acc1, false, false);
    }
    {
      V16U bf;
      bf.o[0] = *(const h8*)(sb + 2 * 16 * 32);
      bf.o[1] = *(const h8*)(sb + 2 * 16 * 32 + 16);
      acc2 = __builtin_amdgcn_wmma_f32_16x16x32_f16(false, af.v, false, bf.v, (short)0, acc2, false, false);
    }
    {
      V16U bf;
      bf.o[0] = *(const h8*)(sb + 3 * 16 * 32);
      bf.o[1] = *(const h8*)(sb + 3 * 16 * 32 + 16);
      acc3 = __builtin_amdgcn_wmma_f32_16x16x32_f16(false, af.v, false, bf.v, (short)0, acc3, false, false);
    }

    ra0 = rb0; ra1 = rb1; ra2 = rb2; ra3 = rb3;

    if (more) {
#if ASYNC_LDS
      __builtin_amdgcn_s_wait_asynccnt(0);
#endif
      __syncthreads();
    }
  }

  // C/D layout: element r of v8f at lane L -> row = r + 8*(L/16), col = L%16
  const long rowBase = (long)blockIdx.y * 128 + wave * 16 + kg * 8;
#pragma unroll
  for (int j = 0; j < 4; ++j) {
    v8f acc = (j == 0) ? acc0 : (j == 1) ? acc1 : (j == 2) ? acc2 : acc3;
    int col = nBase + j * 16 + l15;
    float bv = bias ? bias[col] : 0.0f;
#pragma unroll
    for (int r = 0; r < 8; ++r)
      C[(rowBase + r) * ldc + col] = acc[r] + bv;
  }
}

// ---------------- GRU gate combine ----------------------------------------
__device__ __forceinline__ float sigm(float x) { return 1.0f / (1.0f + __expf(-x)); }

__global__ void k_gru_combine(const float* __restrict__ gx, const float* __restrict__ gh,
                              const float* __restrict__ hprev, float* __restrict__ hnew,
                              float* __restrict__ srcOut, int accumulate) {
  int idx = blockIdx.x * blockDim.x + threadIdx.x;    // B*U
  int b = idx / U_, u = idx - b * U_;
  long g = (long)b * U3_;
  float z = sigm(gx[g + u] + gh[g + u]);
  float r = sigm(gx[g + U_ + u] + gh[g + U_ + u]);
  float c = tanhf(gx[g + 2 * U_ + u] + r * gh[g + 2 * U_ + u]);
  float hn = z * hprev[idx] + (1.0f - z) * c;
  hnew[idx] = hn;
  if (srcOut) {
    long si = (long)b * (S_ * U_) + u;                // srcOut pre-offset by s*U
    if (accumulate) srcOut[si] += hn; else srcOut[si] = hn;
  }
}

// ---------------- attention score: e[b,s] = tanh(sp+hq)·Wav + bav ----------
__global__ void k_attn_score(const float* __restrict__ srcp, const float* __restrict__ hq,
                             const float* __restrict__ Wav, const float* __restrict__ bav,
                             float* __restrict__ e) {
  int lane = threadIdx.x & 31;
  int wid = (blockIdx.x * blockDim.x + threadIdx.x) >> 5;  // 0..B*S-1
  int b = wid >> 5;                                        // S_ == 32
  const float* sp = srcp + (long)wid * U_;
  const float* hp = hq + (long)b * U_;
  float p = 0.0f;
#pragma unroll
  for (int i = 0; i < U_ / 32; ++i) {
    int u = lane + i * 32;
    p += tanhf(sp[u] + hp[u]) * Wav[u];
  }
  for (int off = 16; off; off >>= 1) p += __shfl_xor(p, off, 32);
  if (lane == 0) e[wid] = p + bav[0];
}

// ---------------- softmax over S=32 (one wave per batch row) ---------------
__global__ void k_softmax32(const float* __restrict__ e, float* __restrict__ w) {
  int lane = threadIdx.x & 31;
  int b = (blockIdx.x * blockDim.x + threadIdx.x) >> 5;
  float v = e[b * 32 + lane];
  float m = v;
  for (int off = 16; off; off >>= 1) m = fmaxf(m, __shfl_xor(m, off, 32));
  float ex = __expf(v - m);
  float sum = ex;
  for (int off = 16; off; off >>= 1) sum += __shfl_xor(sum, off, 32);
  w[b * 32 + lane] = ex / sum;
}

// ---------------- weighted sum -> inp[:, D:D+U] ----------------------------
__global__ void k_attend(const float* __restrict__ src, const float* __restrict__ w,
                         float* __restrict__ inp) {
  int idx = blockIdx.x * blockDim.x + threadIdx.x;    // B*U
  int b = idx / U_, u = idx - b * U_;
  const float* sp = src + (long)b * S_ * U_ + u;
  const float* wp = w + b * S_;
  float acc = 0.0f;
#pragma unroll
  for (int s = 0; s < S_; ++s) acc += sp[(long)s * U_] * wp[s];
  inp[(long)b * (D_ + U_) + D_ + u] = acc;
}

__global__ void k_copy_h0(const float* __restrict__ src, float* __restrict__ h) {
  int idx = blockIdx.x * blockDim.x + threadIdx.x;    // B*U
  int b = idx / U_, u = idx - b * U_;
  h[idx] = src[(long)b * S_ * U_ + u];
}

// ---------------------------------------------------------------------------
extern "C" void kernel_launch(void* const* d_in, const int* in_sizes, int n_in,
                              void* d_out, int out_size, void* d_ws, size_t ws_size,
                              hipStream_t stream) {
  const int*   src_ids = (const int*)d_in[0];
  const int*   tgt_ids = (const int*)d_in[1];
  const float* E_s = (const float*)d_in[2];
  const float* W_f = (const float*)d_in[3];
  const float* U_f = (const float*)d_in[4];
  const float* b_f = (const float*)d_in[5];
  const float* W_b = (const float*)d_in[6];
  const float* U_b = (const float*)d_in[7];
  const float* b_b = (const float*)d_in[8];
  const float* E_t = (const float*)d_in[9];
  const float* Wd  = (const float*)d_in[10];
  const float* Ud  = (const float*)d_in[11];
  const float* bd  = (const float*)d_in[12];
  const float* Wo  = (const float*)d_in[13];
  const float* bo  = (const float*)d_in[14];
  const float* Was = (const float*)d_in[15];
  const float* bas = (const float*)d_in[16];
  const float* Wah = (const float*)d_in[17];
  const float* bah = (const float*)d_in[18];
  const float* Wav = (const float*)d_in[19];
  const float* bav = (const float*)d_in[20];
  float* out = (float*)d_out;

  char* ws = (char*)d_ws;
  size_t off = 0;
  auto alloc = [&](size_t bytes) -> char* {
    char* p = ws + off;
    off = (off + bytes + 255) & ~(size_t)255;
    return p;
  };

  _Float16* pWf  = (_Float16*)alloc((size_t)D_ * U3_ * 2);
  _Float16* pUf  = (_Float16*)alloc((size_t)U_ * U3_ * 2);
  _Float16* pWb  = (_Float16*)alloc((size_t)D_ * U3_ * 2);
  _Float16* pUb  = (_Float16*)alloc((size_t)U_ * U3_ * 2);
  _Float16* pWd  = (_Float16*)alloc((size_t)(D_ + U_) * U3_ * 2);
  _Float16* pUd  = (_Float16*)alloc((size_t)U_ * U3_ * 2);
  _Float16* pWo  = (_Float16*)alloc((size_t)U_ * VT_ * 2);
  _Float16* pWas = (_Float16*)alloc((size_t)U_ * U_ * 2);
  _Float16* pWah = (_Float16*)alloc((size_t)U_ * U_ * 2);

  float* xs   = (float*)alloc((size_t)S_ * B_ * D_ * 4);       // [S,B,D]
  float* src  = (float*)alloc((size_t)B_ * S_ * U_ * 4);       // hf + hb, [B,S,U]
  float* srcp = (float*)alloc((size_t)B_ * S_ * U_ * 4);       // src @ Was + bas
  float* hA   = (float*)alloc((size_t)B_ * U_ * 4);
  float* hB   = (float*)alloc((size_t)B_ * U_ * 4);
  float* gx   = (float*)alloc((size_t)B_ * U3_ * 4);
  float* gh   = (float*)alloc((size_t)B_ * U3_ * 4);
  float* hq   = (float*)alloc((size_t)B_ * U_ * 4);
  float* e    = (float*)alloc((size_t)B_ * S_ * 4);
  float* wsm  = (float*)alloc((size_t)B_ * S_ * 4);
  float* inp  = (float*)alloc((size_t)B_ * (D_ + U_) * 4);

  auto pack = [&](const float* W, _Float16* o, int K, int N) {
    int n = K * N;
    k_pack_weight<<<(n + 255) / 256, 256, 0, stream>>>(W, o, K, N);
  };
  auto gemm = [&](const float* A, int lda, const _Float16* Bp, int K,
                  const float* bias, float* C, long ldc, int M, int N) {
    dim3 grid(N / 64, M / 128);
    k_wmma_gemm<<<grid, dim3(256), 0, stream>>>(A, lda, Bp, K, bias, C, ldc);
  };
  auto zero = [&](float* p, int n) {
    k_zero<<<(n + 255) / 256, 256, 0, stream>>>(p, n);
  };
  const int BU = B_ * U_;

  // ---- weight packing (once per launch; ~9.4MB f16, L2-resident after) ----
  pack(W_f, pWf, D_, U3_);   pack(U_f, pUf, U_, U3_);
  pack(W_b, pWb, D_, U3_);   pack(U_b, pUb, U_, U3_);
  pack(Wd,  pWd, D_ + U_, U3_); pack(Ud, pUd, U_, U3_);
  pack(Wo,  pWo, U_, VT_);
  pack(Was, pWas, U_, U_);   pack(Wah, pWah, U_, U_);

  // ---- source embeddings, time-major -------------------------------------
  k_gather_src<<<(B_ * S_ * (D_ / 4) + 255) / 256, 256, 0, stream>>>(src_ids, E_s, xs);

  // ---- forward encoder GRU ------------------------------------------------
  float *hp = hA, *hn = hB;
  zero(hp, BU);
  for (int s = 0; s < S_; ++s) {
    gemm(xs + (size_t)s * B_ * D_, D_, pWf, D_, b_f,        gx, U3_, B_, U3_);
    gemm(hp,                       U_, pUf, U_, b_f + U3_,  gh, U3_, B_, U3_);
    k_gru_combine<<<(BU + 255) / 256, 256, 0, stream>>>(gx, gh, hp, hn,
                                                        src + (size_t)s * U_, 0);
    float* t = hp; hp = hn; hn = t;
  }

  // ---- backward encoder GRU (accumulate into src) -------------------------
  zero(hp, BU);
  for (int i = 0; i < S_; ++i) {
    int s = S_ - 1 - i;
    gemm(xs + (size_t)s * B_ * D_, D_, pWb, D_, b_b,        gx, U3_, B_, U3_);
    gemm(hp,                       U_, pUb, U_, b_b + U3_,  gh, U3_, B_, U3_);
    k_gru_combine<<<(BU + 255) / 256, 256, 0, stream>>>(gx, gh, hp, hn,
                                                        src + (size_t)s * U_, 1);
    float* t = hp; hp = hn; hn = t;
  }

  // ---- src_proj = src @ Was + bas  (65536 x 512 x 512) --------------------
  gemm(src, U_, pWas, U_, bas, srcp, U_, B_ * S_, U_);

  // ---- decoder init: h0 = src[:,0,:]; inp0 = [E_t[BOW], attend(h0)] -------
  k_copy_h0<<<(BU + 255) / 256, 256, 0, stream>>>(src, hp);
  gemm(hp, U_, pWah, U_, bah, hq, U_, B_, U_);
  k_attn_score<<<(B_ * S_ * 32 + 255) / 256, 256, 0, stream>>>(srcp, hq, Wav, bav, e);
  k_softmax32<<<(B_ * 32 + 255) / 256, 256, 0, stream>>>(e, wsm);
  k_attend<<<(BU + 255) / 256, 256, 0, stream>>>(src, wsm, inp);
  k_gather_tgt<<<(B_ * (D_ / 4) + 255) / 256, 256, 0, stream>>>(nullptr, 0, BOW_, E_t, inp);

  // ---- decoder scan -------------------------------------------------------
  for (int t = 0; t < T_; ++t) {
    gemm(inp, D_ + U_, pWd, D_ + U_, bd,       gx, U3_, B_, U3_);
    gemm(hp,  U_,      pUd, U_,      bd + U3_, gh, U3_, B_, U3_);
    k_gru_combine<<<(BU + 255) / 256, 256, 0, stream>>>(gx, gh, hp, hn, nullptr, 0);
    // logits[:, t, :] = h2 @ Wo + bo, strided into d_out [B,T,VT]
    gemm(hn, U_, pWo, U_, bo, out + (size_t)t * VT_, (long)T_ * VT_, B_, VT_);
    if (t + 1 < T_) {
      gemm(hn, U_, pWah, U_, bah, hq, U_, B_, U_);
      k_attn_score<<<(B_ * S_ * 32 + 255) / 256, 256, 0, stream>>>(srcp, hq, Wav, bav, e);
      k_softmax32<<<(B_ * 32 + 255) / 256, 256, 0, stream>>>(e, wsm);
      k_attend<<<(BU + 255) / 256, 256, 0, stream>>>(src, wsm, inp);
      k_gather_tgt<<<(B_ * (D_ / 4) + 255) / 256, 256, 0, stream>>>(tgt_ids, t, -1, E_t, inp);
    }
    float* tmp = hp; hp = hn; hn = tmp;
  }
}
